// VertexPointNet_20736102105289
// MI455X (gfx1250) — 8-bit matrix operand order repaired and checked by reading, compiled, not run on hardware
//
#include <hip/hip_runtime.h>
#include <hip/hip_bf16.h>

typedef __attribute__((ext_vector_type(8)))  float    v8f;
typedef __attribute__((ext_vector_type(4)))  float    v4f;
typedef __attribute__((ext_vector_type(8)))  int      v8i;
typedef __attribute__((ext_vector_type(16))) int      v16i;

#define GSEG   256
#define HD1    64
#define HD2    128
#define LAT    256

#define OFF_W3F 0
#define OFF_W2F 32768
#define OFF_W1  40960
#define OFF_B1  41984
#define OFF_B2  42240
#define OFF_B3  42752
#define HEAP_BYTES 43776

__device__ __forceinline__ unsigned enc_f32(float f) {
    unsigned u = __float_as_uint(f);
    return u ^ ((unsigned)((int)u >> 31) | 0x80000000u);
}
__device__ __forceinline__ float dec_f32(unsigned e) {
    unsigned u = e ^ ((e & 0x80000000u) ? 0x80000000u : 0xFFFFFFFFu);
    return __uint_as_float(u);
}

#if __has_builtin(__builtin_amdgcn_cvt_pk_fp8_f32)
__device__ __forceinline__ int pack4_fp8(float a, float b, float c, float d) {
    int w = 0;
    w = __builtin_amdgcn_cvt_pk_fp8_f32(a, b, w, false);
    w = __builtin_amdgcn_cvt_pk_fp8_f32(c, d, w, true);
    return w;
}
#else
__device__ __forceinline__ unsigned f32_to_fp8(float f) {
    unsigned u = __float_as_uint(f);
    unsigned s = (u >> 24) & 0x80u;
    int e = (int)((u >> 23) & 0xFF) - 127;
    unsigned m = (u >> 20) & 7u;
    if (e < -6) return s;
    if (e > 8)  return s | 0x7Eu;
    return s | (unsigned)((e + 7) << 3) | m;
}
__device__ __forceinline__ int pack4_fp8(float a, float b, float c, float d) {
    return (int)(f32_to_fp8(a) | (f32_to_fp8(b) << 8) |
                 (f32_to_fp8(c) << 16) | (f32_to_fp8(d) << 24));
}
#endif

__device__ __forceinline__ v8f wmma64f8(v8i a, v8i b, v8f c) {
    return __builtin_amdgcn_wmma_f32_16x16x64_fp8_fp8(a, b, (short)0, c, false, false);
}
__device__ __forceinline__ v8f wmma128f8(v16i a, v16i b, v8f c) {
    return __builtin_amdgcn_wmma_f32_16x16x128_fp8_fp8(a, b, (short)0, c, false, false);
}

__global__ void zero_ws_kernel(unsigned* __restrict__ w, int nw) {
    int i = blockIdx.x * blockDim.x + threadIdx.x;
    if (i < nw) w[i] = 0u;
}

__global__ void centroid_kernel(const float* __restrict__ pc,
                                const int* __restrict__ seg,
                                float* __restrict__ csum, int n) {
    __shared__ float sacc[GSEG * 4];
    int tid = threadIdx.x;
    for (int i = tid; i < GSEG * 4; i += blockDim.x) sacc[i] = 0.f;
    __syncthreads();
    int stride = gridDim.x * blockDim.x;
    for (int p = blockIdx.x * blockDim.x + tid; p < n; p += stride) {
        int s = seg[p];
        const float* row = pc + (size_t)p * 5;
        atomicAdd(&sacc[s * 4 + 0], row[1]);
        atomicAdd(&sacc[s * 4 + 1], row[2]);
        atomicAdd(&sacc[s * 4 + 2], row[3]);
        atomicAdd(&sacc[s * 4 + 3], 1.0f);
    }
    __syncthreads();
    for (int i = tid; i < GSEG * 4; i += blockDim.x)
        if (sacc[i] != 0.f) atomicAdd(&csum[i], sacc[i]);
}

__global__ __launch_bounds__(256, 1)
void mlp_pool_kernel(const float* __restrict__ pc,
                     const float* __restrict__ W1, const float* __restrict__ b1,
                     const float* __restrict__ W2, const float* __restrict__ b2,
                     const float* __restrict__ W3, const float* __restrict__ b3,
                     const int* __restrict__ seg,
                     unsigned* __restrict__ gpool, int n) {
    __shared__ alignas(128) unsigned char s_heap[HEAP_BYTES];
    int*   sw3i = (int*)(s_heap + OFF_W3F);
    int*   sw2i = (int*)(s_heap + OFF_W2F);
    float* sW1  = (float*)(s_heap + OFF_W1);
    float* sB1  = (float*)(s_heap + OFF_B1);
    float* sB2  = (float*)(s_heap + OFF_B2);
    float* sB3  = (float*)(s_heap + OFF_B3);

    const int tid  = threadIdx.x;
    const int wv   = tid >> 5;
    const int lane = tid & 31;
    const int hi   = lane >> 4;
    const int ml   = lane & 15;
    const bool lo  = (hi == 0);

    sW1[tid] = W1[tid];
    if (tid < HD1) sB1[tid] = b1[tid];
    if (tid < HD2) sB2[tid] = b2[tid];
    sB3[tid] = b3[tid];

    for (int idx = tid; idx < 16 * 32 * 16; idx += 256) {
        int v  = idx & 15;
        int ln = (idx >> 4) & 31;
        int t  = idx >> 9;
        int q  = ln >> 4;
        int mm = t * 16 + (ln & 15);
        int kb = 4 * (v & 3) + 16 * q + 32 * (v >> 2);
        sw3i[idx] = pack4_fp8(W3[(kb + 0) * LAT + mm], W3[(kb + 1) * LAT + mm],
                              W3[(kb + 2) * LAT + mm], W3[(kb + 3) * LAT + mm]);
    }
    for (int idx = tid; idx < 8 * 32 * 8; idx += 256) {
        int v  = idx & 7;
        int ln = (idx >> 3) & 31;
        int t  = idx >> 8;
        int q  = ln >> 4;
        int mm = t * 16 + (ln & 15);
        int kb = 4 * (v & 3) + 16 * q + 32 * (v >> 2);
        sw2i[idx] = pack4_fp8(W2[(kb + 0) * HD2 + mm], W2[(kb + 1) * HD2 + mm],
                              W2[(kb + 2) * HD2 + mm], W2[(kb + 3) * HD2 + mm]);
    }
    __syncthreads();

    const int*   w3L = sw3i + (lane << 4);
    const int*   w2L = sw2i + (lane << 3);
    const float* W1L = sW1 + 16 * hi;
    const float* b1L = sB1 + 16 * hi;
    const float* b2L = sB2 + 8 * hi;
    const float* b3L = sB3 + 8 * hi;

    float acc[16][8];
#pragma unroll
    for (int t = 0; t < 16; ++t)
#pragma unroll
        for (int g = 0; g < 8; ++g) acc[t][g] = -__builtin_inff();
    int curSeg = -1;

    auto flushAcc = [&]() {
        if (curSeg >= 0) {
#pragma unroll
            for (int t = 0; t < 16; ++t)
#pragma unroll
                for (int g = 0; g < 8; ++g) {
                    float m = acc[t][g];
                    m = __builtin_fmaxf(m, __shfl_xor(m, 1, 32));
                    m = __builtin_fmaxf(m, __shfl_xor(m, 2, 32));
                    m = __builtin_fmaxf(m, __shfl_xor(m, 4, 32));
                    m = __builtin_fmaxf(m, __shfl_xor(m, 8, 32));
                    if (ml == 0)
                        atomicMax(&gpool[(size_t)curSeg * LAT + 16 * t + g + 8 * hi],
                                  enc_f32(m));
                    acc[t][g] = -__builtin_inff();
                }
        }
    };

    const int nChunks = (n + 127) >> 7;
    const int span    = (nChunks + gridDim.x - 1) / gridDim.x;
    const int cBeg    = blockIdx.x * span;
    const int cEnd    = (cBeg + span < nChunks) ? cBeg + span : nChunks;
    if (cBeg >= cEnd) { return; }

    float f0, f1, f2, f3;
    int sA, sB, sP;
    auto loadChunk = [&](int chunk, float& g0, float& g1, float& g2, float& g3,
                         int& a, int& b, int& s) {
        const int bb  = chunk << 7;
        const int pp  = bb + wv * 16 + ml;
        const int ppc = (pp < n - 1) ? pp : (n - 1);
        const float* r = pc + (size_t)ppc * 5;
        g0 = r[4]; g1 = r[1]; g2 = r[2]; g3 = r[3];
        a = seg[bb];
        b = (bb + 127 < n) ? seg[bb + 127] : -2;
        s = seg[ppc];
    };
    loadChunk(cBeg, f0, f1, f2, f3, sA, sB, sP);

    for (int chunk = cBeg; chunk < cEnd; ++chunk) {
        const bool uni = (sA == sB);
        if (uni && sA != curSeg) { flushAcc(); curSeg = sA; }
        const bool valid = ((chunk << 7) + wv * 16 + ml) < n;
        const int  mseg  = sP;

        v8i act1;
#pragma unroll
        for (int c = 0; c < 2; ++c) {
#pragma unroll
            for (int q = 0; q < 4; ++q) {
                v4f bv = *(const v4f*)&b1L[32 * c + 4 * q];
                v4f wa = *(const v4f*)&W1L[32 * c + 4 * q];
                v4f wb = *(const v4f*)&W1L[HD1 + 32 * c + 4 * q];
                v4f wc = *(const v4f*)&W1L[2 * HD1 + 32 * c + 4 * q];
                v4f wd = *(const v4f*)&W1L[3 * HD1 + 32 * c + 4 * q];
                v4f v  = bv + f0 * wa + f1 * wb + f2 * wc + f3 * wd;
                act1[c * 4 + q] = pack4_fp8(__builtin_fmaxf(v[0], 0.f),
                                            __builtin_fmaxf(v[1], 0.f),
                                            __builtin_fmaxf(v[2], 0.f),
                                            __builtin_fmaxf(v[3], 0.f));
            }
        }

        if (chunk + 1 < cEnd)
            loadChunk(chunk + 1, f0, f1, f2, f3, sA, sB, sP);

        v8f h2[8];
        {
            v8i wCur = *(const v8i*)&w2L[0];
            v4f q0 = *(const v4f*)&b2L[0];
            v4f q1 = *(const v4f*)&b2L[4];
            v8f ccP;
#pragma unroll
            for (int t = 0; t < 8; ++t) {
                v8f cc;
#pragma unroll
                for (int e = 0; e < 4; ++e) { cc[e] = q0[e]; cc[4 + e] = q1[e]; }
                v8i wUse = wCur;
                if (t < 7) {
                    wCur = *(const v8i*)&w2L[(t + 1) * 256];
                    q0 = *(const v4f*)&b2L[16 * (t + 1)];
                    q1 = *(const v4f*)&b2L[16 * (t + 1) + 4];
                }
                cc = wmma64f8(wUse, act1, cc);
                if (t > 0) {
#pragma unroll
                    for (int g = 0; g < 8; ++g)
                        h2[t - 1][g] = __builtin_fmaxf(ccP[g], 0.f);
                }
                ccP = cc;
            }
#pragma unroll
            for (int g = 0; g < 8; ++g) h2[7][g] = __builtin_fmaxf(ccP[g], 0.f);
        }

        v16i act3;
#pragma unroll
        for (int grp = 0; grp < 4; ++grp) {
            float xs[8], ys[8];
#pragma unroll
            for (int g = 0; g < 8; ++g) {
                xs[g] = __shfl_xor(h2[2 * grp][g], 16, 32);
                ys[g] = __shfl_xor(h2[2 * grp + 1][g], 16, 32);
            }
            float sel[16];
#pragma unroll
            for (int j = 0; j < 8; ++j)
                sel[j] = lo ? h2[2 * grp][j] : ys[j];
#pragma unroll
            for (int j = 8; j < 16; ++j)
                sel[j] = lo ? xs[j - 8] : h2[2 * grp + 1][j - 8];
#pragma unroll
            for (int v = 0; v < 4; ++v)
                act3[grp * 4 + v] = pack4_fp8(sel[4 * v + 0], sel[4 * v + 1],
                                              sel[4 * v + 2], sel[4 * v + 3]);
        }

        auto proc = [&](v8f cv, int tt) {
            if (uni) {
#pragma unroll
                for (int g = 0; g < 8; ++g)
                    acc[tt][g] = __builtin_fmaxf(acc[tt][g], cv[g]);
            } else if (valid) {
#pragma unroll
                for (int g = 0; g < 8; ++g)
                    atomicMax(&gpool[(size_t)mseg * LAT + 16 * tt + g + 8 * hi],
                              enc_f32(cv[g]));
            }
        };
        {
            v16i wCur = *(const v16i*)&w3L[0];
            v4f p0 = *(const v4f*)&b3L[0];
            v4f p1 = *(const v4f*)&b3L[4];
            v8f ccP;
#pragma unroll
            for (int t = 0; t < 16; ++t) {
                v8f cc;
#pragma unroll
                for (int e = 0; e < 4; ++e) { cc[e] = p0[e]; cc[4 + e] = p1[e]; }
                v16i wUse = wCur;
                if (t < 15) {
                    wCur = *(const v16i*)&w3L[(t + 1) * 512];
                    p0 = *(const v4f*)&b3L[16 * (t + 1)];
                    p1 = *(const v4f*)&b3L[16 * (t + 1) + 4];
                }
                cc = wmma128f8(wUse, act3, cc);
                if (t > 0) proc(ccP, t - 1);
                ccP = cc;
            }
            proc(ccP, 15);
        }
    }
    flushAcc();
}

__global__ void finalize_kernel(const float* __restrict__ Wf,
                                const float* __restrict__ bf,
                                const float* __restrict__ csum,
                                const unsigned* __restrict__ gpool,
                                float* __restrict__ out) {
    int s = threadIdx.x;
    float cnt = fmaxf(csum[s * 4 + 3], 1.0f);
    float c0 = csum[s * 4 + 0] / cnt;
    float c1 = csum[s * 4 + 1] / cnt;
    float c2 = csum[s * 4 + 2] / cnt;
    float a0 = bf[0], a1 = bf[1], a2 = bf[2];
    for (int d = 0; d < LAT; ++d) {
        float v = dec_f32(gpool[s * LAT + d]);
        a0 += v * Wf[d * 3 + 0];
        a1 += v * Wf[d * 3 + 1];
        a2 += v * Wf[d * 3 + 2];
    }
    auto sp = [](float x) { return fmaxf(x, 0.f) + log1pf(expf(-fabsf(x))); };
    out[s * 3 + 0] = c0 + sp(a0);
    out[s * 3 + 1] = c1 + sp(a1);
    out[s * 3 + 2] = c2 + sp(a2);
}

extern "C" void kernel_launch(void* const* d_in, const int* in_sizes, int n_in,
                              void* d_out, int out_size, void* d_ws, size_t ws_size,
                              hipStream_t stream) {
    const float* pc = (const float*)d_in[0];
    const float* W1 = (const float*)d_in[1];
    const float* b1 = (const float*)d_in[2];
    const float* W2 = (const float*)d_in[3];
    const float* b2 = (const float*)d_in[4];
    const float* W3 = (const float*)d_in[5];
    const float* b3 = (const float*)d_in[6];
    const float* Wf = (const float*)d_in[7];
    const float* bf = (const float*)d_in[8];
    const int*  seg = (const int*)d_in[9];
    const int n = in_sizes[0] / 5;

    float*    csum  = (float*)d_ws;
    unsigned* gpool = (unsigned*)((char*)d_ws + 4096);

    const int nwords = GSEG * 4 + GSEG * LAT;
    zero_ws_kernel<<<(nwords + 255) / 256, 256, 0, stream>>>((unsigned*)d_ws, nwords);
    centroid_kernel<<<512, 256, 0, stream>>>(pc, seg, csum, n);
    mlp_pool_kernel<<<512, 256, 0, stream>>>(pc, W1, b1, W2, b2, W3, b3, seg, gpool, n);
    finalize_kernel<<<1, 256, 0, stream>>>(Wf, bf, csum, gpool, (float*)d_out);
}
